// MoEReadout_49950469652580
// MI455X (gfx1250) — matrix operand order repaired and checked by reading, not compiled, not run on hardware
//
#include <hip/hip_runtime.h>
#include <hip/hip_bf16.h>
#include <cstdint>

// ---------------- problem constants ----------------
#define IN_F     512
#define HID      512
#define OUT_F    256
#define EMB_DIM  16
#define NR       6      // routed experts
#define NE_TOT   8      // 6 routed + 2 shared
#define TILE_M   64     // atoms per block
#define NTHREADS 512    // 16 waves
#define NWAVES   16
#define PITCH    520    // LDS row pitch in halves (512 + 8 pad -> conflict-free)

typedef _Float16 h8  __attribute__((ext_vector_type(8)));
typedef _Float16 h16 __attribute__((ext_vector_type(16)));
typedef float    f8  __attribute__((ext_vector_type(8)));

// ---------------- WMMA tile loaders (same contiguous k order in A and B) ----------------
// A (16-bit, 16x32): lane l -> M = l&15; element e -> K = 16*hi + e
__device__ __forceinline__ h16 load_a_lds(const _Float16* buf, int mtile, int kb, int lr, int hi) {
    const _Float16* row = buf + (mtile * 16 + lr) * PITCH;
    h8 x = *(const h8*)(row + kb + hi * 16);
    h8 y = *(const h8*)(row + kb + hi * 16 + 8);
    return __builtin_shufflevector(x, y, 0,1,2,3,4,5,6,7,8,9,10,11,12,13,14,15);
}
// B (16-bit, 32x16): lane l -> N = l&15; element e -> K = 16*hi + e.
// Weights stored row-major [N][K], 16 consecutive halves along K per lane.
__device__ __forceinline__ h16 load_b_glb(const _Float16* __restrict__ w, int ld,
                                          int nb, int kb, int lr, int hi) {
    return *(const h16*)(w + (size_t)(nb + lr) * ld + kb + hi * 16);
}

__device__ __forceinline__ f8 wmma_f16(h16 a, h16 b, f8 c) {
    return __builtin_amdgcn_wmma_f32_16x16x32_f16(false, a, false, b, (short)0, c, false, false);
}

__device__ __forceinline__ float silu(float x) { return x / (1.0f + __expf(-x)); }

// CDNA5 async global->LDS copy, 16 bytes per lane, tracked by ASYNCcnt.
__device__ __forceinline__ void async_copy_b128(uint32_t lds_off, const void* gaddr) {
    asm volatile("global_load_async_to_lds_b128 %0, %1, off"
                 :: "v"(lds_off), "v"(gaddr) : "memory");
}
__device__ __forceinline__ void wait_asynccnt0() {
    asm volatile("s_wait_asynccnt 0x0" ::: "memory");
}

// ---------------- pass 0a: fp32 -> f16 weight conversion ----------------
__global__ void convert_kernel(const float* __restrict__ rW1, const float* __restrict__ sW1,
                               const float* __restrict__ rW2, const float* __restrict__ sW2,
                               const float* __restrict__ rb1, const float* __restrict__ sb1,
                               const float* __restrict__ rb2, const float* __restrict__ sb2,
                               _Float16* __restrict__ W1h, _Float16* __restrict__ W2h,
                               float* __restrict__ b1f, float* __restrict__ b2f) {
    const long NW1 = (long)NE_TOT * HID * IN_F;
    const long NW2 = (long)NE_TOT * OUT_F * HID;
    const long NB1 = (long)NE_TOT * HID;
    const long NB2 = (long)NE_TOT * OUT_F;
    long t = (long)blockIdx.x * blockDim.x + threadIdx.x;
    if (t < NW1) {
        long e = t / ((long)HID * IN_F), r = t % ((long)HID * IN_F);
        float v = (e < NR) ? rW1[e * (long)HID * IN_F + r] : sW1[(e - NR) * (long)HID * IN_F + r];
        W1h[t] = (_Float16)v;
        return;
    }
    t -= NW1;
    if (t < NW2) {
        long e = t / ((long)OUT_F * HID), r = t % ((long)OUT_F * HID);
        float v = (e < NR) ? rW2[e * (long)OUT_F * HID + r] : sW2[(e - NR) * (long)OUT_F * HID + r];
        W2h[t] = (_Float16)v;
        return;
    }
    t -= NW2;
    if (t < NB1) {
        long e = t / HID, r = t % HID;
        b1f[t] = (e < NR) ? rb1[e * HID + r] : sb1[(e - NR) * HID + r];
        return;
    }
    t -= NB1;
    if (t < NB2) {
        long e = t / OUT_F, r = t % OUT_F;
        b2f[t] = (e < NR) ? rb2[e * OUT_F + r] : sb2[(e - NR) * OUT_F + r];
    }
}

// ---------------- pass 0b: fp32 features -> f16 (enables async gather) ----------------
__global__ void feat_convert_kernel(const float* __restrict__ features,
                                    _Float16* __restrict__ featH, long total4) {
    long t = (long)blockIdx.x * blockDim.x + threadIdx.x;
    if (t >= total4) return;
    const float4 v = *(const float4*)(features + t * 4);
    _Float16* d = featH + t * 4;
    d[0] = (_Float16)v.x; d[1] = (_Float16)v.y; d[2] = (_Float16)v.z; d[3] = (_Float16)v.w;
}

// ---------------- pass 1: router + expert bucketing ----------------
__global__ void router_kernel(const int* __restrict__ species, const float* __restrict__ emb,
                              const float* __restrict__ Wr,
                              int* __restrict__ cnt, int* __restrict__ idxL, float* __restrict__ wL,
                              int n) {
    int i = blockIdx.x * blockDim.x + threadIdx.x;
    if (i >= n) return;
    int z = species[i];
    float u[EMB_DIM];
#pragma unroll
    for (int d = 0; d < EMB_DIM; ++d) { float x = emb[z * EMB_DIM + d]; u[d] = silu(x); }
    float sc[NR]; float mx = -1e30f;
#pragma unroll
    for (int j = 0; j < NR; ++j) {
        float s = 0.0f;
#pragma unroll
        for (int d = 0; d < EMB_DIM; ++d) s += u[d] * Wr[j * EMB_DIM + d];
        sc[j] = s; mx = fmaxf(mx, s);
    }
    float denom = 0.0f;
#pragma unroll
    for (int j = 0; j < NR; ++j) { sc[j] = __expf(sc[j] - mx); denom += sc[j]; }
    float inv = 1.0f / denom;
#pragma unroll
    for (int j = 0; j < NR; ++j) sc[j] *= inv;
    // top-2 with lower-index tie-break (matches jax.lax.top_k)
    int a = 0;
#pragma unroll
    for (int j = 1; j < NR; ++j) if (sc[j] > sc[a]) a = j;
    int b = (a == 0) ? 1 : 0;
#pragma unroll
    for (int j = 0; j < NR; ++j) { if (j == a || j == b) continue; if (sc[j] > sc[b]) b = j; }
    int pa = atomicAdd(&cnt[a], 1);
    idxL[(size_t)a * n + pa] = i; wL[(size_t)a * n + pa] = sc[a];
    int pb = atomicAdd(&cnt[b], 1);
    idxL[(size_t)b * n + pb] = i; wL[(size_t)b * n + pb] = sc[b];
    // shared experts: identity lists, weight 1 (deterministic slots)
    idxL[(size_t)6 * n + i] = i; wL[(size_t)6 * n + i] = 1.0f;
    idxL[(size_t)7 * n + i] = i; wL[(size_t)7 * n + i] = 1.0f;
}

// ---------------- pass 2: fused expert MLP ----------------
// gather -> WMMA GEMM1 -> SiLU -> WMMA GEMM2 -> scatter-add
template <bool USE_F16_FEAT>
__global__ __launch_bounds__(NTHREADS, 1)
void moe_expert_kernel(const float* __restrict__ features,
                       const _Float16* __restrict__ featH,
                       const _Float16* __restrict__ W1h, const _Float16* __restrict__ W2h,
                       const float* __restrict__ b1f, const float* __restrict__ b2f,
                       const int* __restrict__ cnt, const int* __restrict__ idxL,
                       const float* __restrict__ wL,
                       float* __restrict__ out, int n, int tilesPerExpert) {
    __shared__ _Float16 feat[TILE_M * PITCH];   // 64 x 512 f16 feature tile
    __shared__ _Float16 hbuf[TILE_M * PITCH];   // 64 x 512 f16 hidden tile
    __shared__ int   s_idx[TILE_M];
    __shared__ float s_w[TILE_M];

    int e    = blockIdx.x / tilesPerExpert;
    int tile = blockIdx.x % tilesPerExpert;
    int count = (e >= NR) ? n : cnt[e];
    int base  = tile * TILE_M;
    if (base >= count) return;

    int tid = threadIdx.x;
    if (tid < TILE_M) {
        int p = base + tid;
        if (p < count) { s_idx[tid] = idxL[(size_t)e * n + p]; s_w[tid] = wL[(size_t)e * n + p]; }
        else           { s_idx[tid] = 0;                        s_w[tid] = 0.0f; }
    }
    __syncthreads();

    if (USE_F16_FEAT) {
        // async DMA gather: 16B per lane per op, ASYNCcnt-tracked
        for (int i = tid * 8; i < TILE_M * IN_F; i += NTHREADS * 8) {
            int r = i >> 9;            // /512
            int c = i & (IN_F - 1);
            uint32_t lds_off = (uint32_t)(uintptr_t)(&feat[r * PITCH + c]);
            const void* g = (const void*)(featH + (size_t)s_idx[r] * IN_F + c);
            async_copy_b128(lds_off, g);
        }
        wait_asynccnt0();
    } else {
        // fallback: fp32 gather + convert in VALU
        for (int i = tid * 4; i < TILE_M * IN_F; i += NTHREADS * 4) {
            int r = i >> 9;
            int c = i & (IN_F - 1);
            const float4 v = *(const float4*)(features + (size_t)s_idx[r] * IN_F + c);
            _Float16* d = &feat[r * PITCH + c];
            d[0] = (_Float16)v.x; d[1] = (_Float16)v.y; d[2] = (_Float16)v.z; d[3] = (_Float16)v.w;
        }
    }
    __syncthreads();

    const int lane = tid & 31, wave = tid >> 5;
    const int hi = lane >> 4, lr = lane & 15;

    // ---- GEMM1: h[64x512] = feat[64x512] * W1^T; per-wave 2 N-tiles x 4 M-tiles ----
    const _Float16* w1 = W1h + (size_t)e * HID * IN_F;
    const float*    b1 = b1f + e * HID;
    f8 acc[2][4] = {};
    for (int kb = 0; kb < IN_F; kb += 32) {
        h16 a0 = load_a_lds(feat, 0, kb, lr, hi);
        h16 a1 = load_a_lds(feat, 1, kb, lr, hi);
        h16 a2 = load_a_lds(feat, 2, kb, lr, hi);
        h16 a3 = load_a_lds(feat, 3, kb, lr, hi);
#pragma unroll
        for (int j = 0; j < 2; ++j) {
            h16 b = load_b_glb(w1, IN_F, (wave * 2 + j) * 16, kb, lr, hi);
            acc[j][0] = wmma_f16(a0, b, acc[j][0]);
            acc[j][1] = wmma_f16(a1, b, acc[j][1]);
            acc[j][2] = wmma_f16(a2, b, acc[j][2]);
            acc[j][3] = wmma_f16(a3, b, acc[j][3]);
        }
    }
    // bias + SiLU -> f16 hidden tile
#pragma unroll
    for (int j = 0; j < 2; ++j) {
        int nc = (wave * 2 + j) * 16 + lr;
        float bias = b1[nc];
#pragma unroll
        for (int m = 0; m < 4; ++m)
#pragma unroll
            for (int r = 0; r < 8; ++r) {
                int row = m * 16 + hi * 8 + r;
                hbuf[row * PITCH + nc] = (_Float16)silu(acc[j][m][r] + bias);
            }
    }
    __syncthreads();

    // ---- GEMM2: o[64x256] = h[64x512] * W2^T; per-wave 1 N-tile x 4 M-tiles ----
    const _Float16* w2 = W2h + (size_t)e * OUT_F * HID;
    const float*    b2 = b2f + e * OUT_F;
    f8 acc2[4] = {};
    for (int kb = 0; kb < HID; kb += 32) {
        h16 a0 = load_a_lds(hbuf, 0, kb, lr, hi);
        h16 a1 = load_a_lds(hbuf, 1, kb, lr, hi);
        h16 a2 = load_a_lds(hbuf, 2, kb, lr, hi);
        h16 a3 = load_a_lds(hbuf, 3, kb, lr, hi);
        h16 b = load_b_glb(w2, HID, wave * 16, kb, lr, hi);
        acc2[0] = wmma_f16(a0, b, acc2[0]);
        acc2[1] = wmma_f16(a1, b, acc2[1]);
        acc2[2] = wmma_f16(a2, b, acc2[2]);
        acc2[3] = wmma_f16(a3, b, acc2[3]);
    }
    // epilogue: weight * (acc + b2) scatter-add into out
    {
        int nc = wave * 16 + lr;
        float bias = b2[nc];
#pragma unroll
        for (int m = 0; m < 4; ++m)
#pragma unroll
            for (int r = 0; r < 8; ++r) {
                int row = m * 16 + hi * 8 + r;
                float wgt = s_w[row];
                if (wgt != 0.0f) {
                    float v = wgt * (acc2[m][r] + bias);
                    atomicAdd(out + (size_t)s_idx[row] * OUT_F + nc, v);
                }
            }
    }
}

// ---------------- host launcher ----------------
extern "C" void kernel_launch(void* const* d_in, const int* in_sizes, int n_in,
                              void* d_out, int out_size, void* d_ws, size_t ws_size,
                              hipStream_t stream) {
    const float* features = (const float*)d_in[0];
    const int*   species  = (const int*)d_in[1];
    const float* emb      = (const float*)d_in[2];
    const float* Wr       = (const float*)d_in[3];
    const float* rW1      = (const float*)d_in[4];
    const float* rb1      = (const float*)d_in[5];
    const float* rW2      = (const float*)d_in[6];
    const float* rb2      = (const float*)d_in[7];
    const float* sW1      = (const float*)d_in[8];
    const float* sb1      = (const float*)d_in[9];
    const float* sW2      = (const float*)d_in[10];
    const float* sb2      = (const float*)d_in[11];
    float* out = (float*)d_out;
    const int n = in_sizes[0] / IN_F;

    // workspace carve-out (256B aligned)
    char* ws = (char*)d_ws;
    size_t off = 0;
    auto take = [&](size_t bytes) { size_t o = off; off = (off + bytes + 255) & ~(size_t)255; return o; };
    int*      cnt  = (int*)(ws + take(NE_TOT * sizeof(int)));
    int*      idxL = (int*)(ws + take((size_t)NE_TOT * n * sizeof(int)));
    float*    wL   = (float*)(ws + take((size_t)NE_TOT * n * sizeof(float)));
    _Float16* W1h  = (_Float16*)(ws + take((size_t)NE_TOT * HID * IN_F * sizeof(_Float16)));
    _Float16* W2h  = (_Float16*)(ws + take((size_t)NE_TOT * OUT_F * HID * sizeof(_Float16)));
    float*    b1f  = (float*)(ws + take(NE_TOT * HID * sizeof(float)));
    float*    b2f  = (float*)(ws + take(NE_TOT * OUT_F * sizeof(float)));
    size_t featOff = take((size_t)n * IN_F * sizeof(_Float16));
    _Float16* featH = (_Float16*)(ws + featOff);
    const bool useF16 = (off <= ws_size);   // f16 feature mirror fits?
    (void)n_in; (void)out_size;

    hipMemsetAsync(cnt, 0, NE_TOT * sizeof(int), stream);
    hipMemsetAsync(out, 0, (size_t)n * OUT_F * sizeof(float), stream);

    const long totalConv = (long)NE_TOT * HID * IN_F + (long)NE_TOT * OUT_F * HID
                         + (long)NE_TOT * HID + (long)NE_TOT * OUT_F;
    convert_kernel<<<(int)((totalConv + 255) / 256), 256, 0, stream>>>(
        rW1, sW1, rW2, sW2, rb1, sb1, rb2, sb2, W1h, W2h, b1f, b2f);

    if (useF16) {
        long total4 = (long)n * IN_F / 4;
        feat_convert_kernel<<<(int)((total4 + 255) / 256), 256, 0, stream>>>(features, featH, total4);
    }

    router_kernel<<<(n + 255) / 256, 256, 0, stream>>>(species, emb, Wr, cnt, idxL, wL, n);

    const int tiles = (n + TILE_M - 1) / TILE_M;
    if (useF16) {
        moe_expert_kernel<true><<<NE_TOT * tiles, NTHREADS, 0, stream>>>(
            features, featH, W1h, W2h, b1f, b2f, cnt, idxL, wL, out, n, tiles);
    } else {
        moe_expert_kernel<false><<<NE_TOT * tiles, NTHREADS, 0, stream>>>(
            features, featH, W1h, W2h, b1f, b2f, cnt, idxL, wL, out, n, tiles);
    }
}
